// Pooling_3D_Layer_34093450395753
// MI455X (gfx1250) — matrix operand order repaired and checked by reading, not compiled, not run on hardware
//
#include <hip/hip_runtime.h>

#define NF 300000
#define NC 60000
#define NE 300000
#define NM 360000   // NF + NC

typedef __bf16 bf16_t;
typedef __attribute__((ext_vector_type(16))) __bf16 v16bf;
typedef __attribute__((ext_vector_type(8)))  float  v8f;

union FragBF { v16bf v; uint4 q[2]; };

__constant__ float c_sig[15] = {
    1.0f, 1.5f, 2.25f, 3.375f, 5.0625f, 7.59375f, 11.390625f,
    17.0859375f, 25.62890625f, 38.443359375f, 57.6650390625f,
    86.49755859375f, 129.746337890625f, 194.6195068359375f,
    291.92926025390625f };

// ---------------------------------------------------------------------------
// Weight prep: W [K,128] f32 -> Wt [128, Kpad] bf16 (zero pad K..Kpad)
// ---------------------------------------------------------------------------
__global__ void convert_wt_kernel(const float* __restrict__ W,
                                  bf16_t* __restrict__ Wt, int K, int Kpad) {
    int idx = blockIdx.x * blockDim.x + threadIdx.x;
    if (idx >= 128 * Kpad) return;
    int n = idx / Kpad, k = idx - n * Kpad;
    float v = (k < K) ? W[(size_t)k * 128 + n] : 0.0f;
    Wt[(size_t)n * Kpad + k] = (bf16_t)v;
}

// ---------------------------------------------------------------------------
// Edge input gather: [fine_h[src] | coarse_h[dst-NF] | efeat | rbf | pad] bf16
// ---------------------------------------------------------------------------
__global__ void edge_gather_kernel(const float* __restrict__ fine_h,
                                   const float* __restrict__ coarse_h,
                                   const float* __restrict__ fine_x,
                                   const float* __restrict__ pool_x,
                                   const float* __restrict__ efeat,
                                   const int* __restrict__ src,
                                   const int* __restrict__ dst,
                                   bf16_t* __restrict__ edge_in, int E) {
    size_t idx = (size_t)blockIdx.x * blockDim.x + threadIdx.x;
    if (idx >= (size_t)E * 416) return;
    int e = (int)(idx / 416);
    int k = (int)(idx - (size_t)e * 416);
    int s = src[e], d = dst[e];
    float val;
    if (k < 128) {
        val = fine_h[(size_t)s * 128 + k];
    } else if (k < 256) {
        val = coarse_h[(size_t)(d - NF) * 128 + (k - 128)];
    } else if (k < 288) {
        val = efeat[(size_t)e * 32 + (k - 256)];
    } else if (k < 303) {
        float dx = fine_x[s * 3 + 0] - pool_x[d * 3 + 0];
        float dy = fine_x[s * 3 + 1] - pool_x[d * 3 + 1];
        float dz = fine_x[s * 3 + 2] - pool_x[d * 3 + 2];
        float d2 = dx * dx + dy * dy + dz * dz;
        val = __expf(-d2 / c_sig[k - 288]);
    } else {
        val = 0.0f;
    }
    edge_in[idx] = (bf16_t)val;
}

__global__ void xrel_kernel(const float* __restrict__ pool_x,
                            const int* __restrict__ src,
                            const int* __restrict__ dst,
                            float* __restrict__ x_rel, int E) {
    int i = blockIdx.x * blockDim.x + threadIdx.x;
    if (i >= E * 3) return;
    int e = i / 3, j = i - e * 3;
    x_rel[i] = pool_x[src[e] * 3 + j] - pool_x[dst[e] * 3 + j];
}

// ---------------------------------------------------------------------------
// WMMA GEMM + fused bias + LayerNorm + optional LeakyReLU (+ optional skip).
// A [M, KPAD] bf16, Wt [128, KPAD] bf16, N = 128.
// Block: 256 threads = 8 waves; wave -> 16 rows x 128 cols (8 wmma tiles).
// Double-buffered 8KB weight slices in LDS (1 barrier per k-step); epilogue
// stages the LN'd tile per wave in LDS and writes coalesced b128 stores.
// ---------------------------------------------------------------------------
template <int KPAD, bool F32OUT>
__global__ __launch_bounds__(256)
void gemm_ln_kernel(const bf16_t* __restrict__ A, int M,
                    const bf16_t* __restrict__ Wt,
                    const float* __restrict__ bias,
                    const float* __restrict__ gamma,
                    const float* __restrict__ beta,
                    int do_lrelu,
                    bf16_t* __restrict__ out_bf,
                    float* __restrict__ out_f32,
                    const float* __restrict__ skip_src,
                    float skip_w) {
    constexpr int SLICE = 128 * 32 * 2;                         // 8KB / slice
    constexpr int EPI   = F32OUT ? 8 * 16 * 128 * 4 : 8 * 16 * 128 * 2;
    constexpr int SMEM  = (2 * SLICE > EPI) ? 2 * SLICE : EPI;
    __shared__ __align__(16) char smem[SMEM];
    bf16_t* lB0 = (bf16_t*)smem;
    bf16_t* lB1 = (bf16_t*)(smem + SLICE);

    const int tid  = threadIdx.x;
    const int wave = tid >> 5;
    const int lane = tid & 31;
    const int hlf  = lane >> 4;
    const int nl   = lane & 15;
    const size_t row0 = (size_t)blockIdx.x * 128 + (size_t)wave * 16;
    const size_t rowA = row0 + nl;              // A-frag row this lane loads
    const int kbase = hlf * 8;                  // 16-bit A layout K sub-offset
    const bool avalid = rowA < (size_t)M;

    // stage Wt[:, k0:k0+32] into buf: each thread copies 32B
    const int stg_n  = tid >> 1;
    const int stg_kh = (tid & 1) * 16;
    auto stage = [&](bf16_t* buf, int k0) {
        const uint4* s = (const uint4*)(Wt + (size_t)stg_n * KPAD + k0 + stg_kh);
        uint4* d = (uint4*)(buf + stg_n * 32 + stg_kh);
        d[0] = s[0];
        d[1] = s[1];
    };

    v8f acc[8] = {};

    stage(lB0, 0);
    __syncthreads();

    constexpr int NSTEP = KPAD / 32;
#pragma unroll
    for (int s = 0; s < NSTEP; ++s) {
        bf16_t* cur = (s & 1) ? lB1 : lB0;
        bf16_t* nxt = (s & 1) ? lB0 : lB1;
        if (s + 1 < NSTEP) stage(nxt, (s + 1) * 32);

        const int k0 = s * 32;
        FragBF fa;
        if (avalid) {
            const bf16_t* pa = A + rowA * (size_t)KPAD + k0 + kbase;
            fa.q[0] = *(const uint4*)(pa);        // K = kbase .. kbase+7
            fa.q[1] = *(const uint4*)(pa + 16);   // K = kbase+16 .. kbase+23
        } else {
            fa.q[0] = make_uint4(0, 0, 0, 0);
            fa.q[1] = make_uint4(0, 0, 0, 0);
        }
#pragma unroll
        for (int t = 0; t < 8; ++t) {
            FragBF fb;   // B layout: col = nl, K = hlf*8 .. +7 and 16+hlf*8 .. +7
            const uint4* pb = (const uint4*)(cur + (t * 16 + nl) * 32 + hlf * 8);
            fb.q[0] = pb[0];
            fb.q[1] = pb[2];
            acc[t] = __builtin_amdgcn_wmma_f32_16x16x32_bf16(
                false, fa.v, false, fb.v, (short)0, acc[t], false, false);
        }
        __syncthreads();   // separates reads of cur from next overwrite
    }

    // ---- Epilogue: bias + LN (+LReLU), stage tile in per-wave LDS region ----
    float bcol[8], gcol[8], btcol[8];
#pragma unroll
    for (int t = 0; t < 8; ++t) {
        int c = t * 16 + nl;
        bcol[t]  = bias[c];
        gcol[t]  = gamma[c];
        btcol[t] = beta[c];
    }

    float* epf  = (float*)smem + (size_t)wave * 16 * 128;
    bf16_t* epb = (bf16_t*)smem + (size_t)wave * 16 * 128;

#pragma unroll
    for (int v = 0; v < 8; ++v) {
        float x[8], s = 0.0f, sq = 0.0f;
#pragma unroll
        for (int t = 0; t < 8; ++t) {
            x[t] = acc[t][v] + bcol[t];
            s  += x[t];
            sq += x[t] * x[t];
        }
#pragma unroll
        for (int m = 1; m < 16; m <<= 1) {   // row lives in one 16-lane half
            s  += __shfl_xor(s,  m, 32);
            sq += __shfl_xor(sq, m, 32);
        }
        float mean = s * (1.0f / 128.0f);
        float var  = sq * (1.0f / 128.0f) - mean * mean;
        float rstd = rsqrtf(var + 1e-5f);
        int rr = hlf * 8 + v;                // row within wave tile
#pragma unroll
        for (int t = 0; t < 8; ++t) {
            float y = (x[t] - mean) * rstd * gcol[t] + btcol[t];
            if (do_lrelu) y = (y > 0.0f) ? y : 0.01f * y;
            int c = t * 16 + nl;
            if (F32OUT) epf[rr * 128 + c] = y;
            else        epb[rr * 128 + c] = (bf16_t)y;
        }
    }
    // same-wave DS ops are ordered; no barrier needed before readback.

    // ---- Coalesced write-out: lane covers half a row (64 elements) ----
    {
        int rr = lane >> 1;                  // 0..15
        int hh = lane & 1;                   // which half of the row
        size_t grow = row0 + rr;
        if (grow < (size_t)M) {
            if (F32OUT) {
                const float* sp = epf + rr * 128 + hh * 64;
                float* dst = out_f32 + grow * 128 + hh * 64;
                const float* sk = skip_src ? (skip_src + grow * 128 + hh * 64)
                                           : nullptr;
#pragma unroll
                for (int i = 0; i < 16; ++i) {
                    float4 vy = ((const float4*)sp)[i];
                    if (sk) {
                        float4 vk = ((const float4*)sk)[i];
                        vy.x = skip_w * vy.x + (1.0f - skip_w) * vk.x;
                        vy.y = skip_w * vy.y + (1.0f - skip_w) * vk.y;
                        vy.z = skip_w * vy.z + (1.0f - skip_w) * vk.z;
                        vy.w = skip_w * vy.w + (1.0f - skip_w) * vk.w;
                    }
                    ((float4*)dst)[i] = vy;
                }
            } else {
                const uint4* sp = (const uint4*)(epb + rr * 128 + hh * 64);
                uint4* dst = (uint4*)(out_bf + grow * 128 + hh * 64);
#pragma unroll
                for (int i = 0; i < 8; ++i) dst[i] = sp[i];
            }
        }
    }
}

// ---------------------------------------------------------------------------
// coords final dot (128 -> 1) + coordinate scatter-mean accumulation
// ---------------------------------------------------------------------------
__global__ __launch_bounds__(256)
void coef_xaggr_kernel(const bf16_t* __restrict__ ch1,
                       const float* __restrict__ cW2,
                       const float* __restrict__ cb2,
                       const float* __restrict__ x_rel,
                       const int* __restrict__ dst,
                       float* __restrict__ xsum,
                       float* __restrict__ cnt, int E) {
    __shared__ float sW[128];
    __shared__ float sb;
    if (threadIdx.x < 128) sW[threadIdx.x] = cW2[threadIdx.x];
    if (threadIdx.x == 0)  sb = cb2[0];
    __syncthreads();
    int e = blockIdx.x * blockDim.x + threadIdx.x;
    if (e >= E) return;
    const bf16_t* row = ch1 + (size_t)e * 128;
    float a = sb;
#pragma unroll 8
    for (int k = 0; k < 128; ++k) a += (float)row[k] * sW[k];
    int d = dst[e];
    atomicAdd(&xsum[d * 3 + 0], x_rel[e * 3 + 0] * a);
    atomicAdd(&xsum[d * 3 + 1], x_rel[e * 3 + 1] * a);
    atomicAdd(&xsum[d * 3 + 2], x_rel[e * 3 + 2] * a);
    atomicAdd(&cnt[d], 1.0f);
}

__global__ void msg_aggr_kernel(const bf16_t* __restrict__ msg,
                                const int* __restrict__ dst,
                                float* __restrict__ msum, int E) {
    size_t i = (size_t)blockIdx.x * blockDim.x + threadIdx.x;
    if (i >= (size_t)E * 128) return;
    int e = (int)(i >> 7), c = (int)(i & 127);
    atomicAdd(&msum[(size_t)dst[e] * 128 + c], (float)msg[i]);
}

__global__ void node_prep_kernel(const float* __restrict__ pool_h,
                                 const float* __restrict__ msum,
                                 const float* __restrict__ cnt,
                                 bf16_t* __restrict__ node_in, int M) {
    size_t i = (size_t)blockIdx.x * blockDim.x + threadIdx.x;
    if (i >= (size_t)M * 256) return;
    int m = (int)(i >> 8), c = (int)(i & 255);
    float v = (c < 128)
                  ? pool_h[(size_t)m * 128 + c]
                  : msum[(size_t)m * 128 + (c - 128)] / fmaxf(cnt[m], 1.0f);
    node_in[i] = (bf16_t)v;
}

__global__ void xout_kernel(const float* __restrict__ og,
                            const float* __restrict__ px,
                            const float* __restrict__ xsum,
                            const float* __restrict__ cnt,
                            float* __restrict__ xout, int M) {
    int i = blockIdx.x * blockDim.x + threadIdx.x;
    if (i >= M * 3) return;
    int m = i / 3;
    xout[i] = 0.25f * og[i] + 0.75f * px[i] + xsum[i] / fmaxf(cnt[m], 1.0f);
}

// ---------------------------------------------------------------------------
extern "C" void kernel_launch(void* const* d_in, const int* in_sizes, int n_in,
                              void* d_out, int out_size, void* d_ws, size_t ws_size,
                              hipStream_t stream) {
    (void)in_sizes; (void)n_in; (void)out_size; (void)ws_size;

    const float* fine_h[2]   = {(const float*)d_in[0],  (const float*)d_in[1]};
    const float* coarse_h[2] = {(const float*)d_in[2],  (const float*)d_in[3]};
    const float* fine_x[2]   = {(const float*)d_in[4],  (const float*)d_in[5]};
    const float* pool_h[2]   = {(const float*)d_in[6],  (const float*)d_in[7]};
    const float* pool_x[2]   = {(const float*)d_in[8],  (const float*)d_in[9]};
    const float* og_x[2]     = {(const float*)d_in[10], (const float*)d_in[11]};
    const float* efeat[2]    = {(const float*)d_in[12], (const float*)d_in[13]};
    const int*   esrc[2]     = {(const int*)d_in[14],   (const int*)d_in[16]};
    const int*   edst[2]     = {(const int*)d_in[15],   (const int*)d_in[17]};
    // params in dict order
    const float* eW1  = (const float*)d_in[18];
    const float* eb1  = (const float*)d_in[19];
    const float* eg1  = (const float*)d_in[20];
    const float* ebt1 = (const float*)d_in[21];
    const float* eW2  = (const float*)d_in[22];
    const float* eb2  = (const float*)d_in[23];
    const float* eg2  = (const float*)d_in[24];
    const float* ebt2 = (const float*)d_in[25];
    const float* nW1  = (const float*)d_in[26];
    const float* nb1  = (const float*)d_in[27];
    const float* ng1  = (const float*)d_in[28];
    const float* nbt1 = (const float*)d_in[29];
    const float* nW2  = (const float*)d_in[30];
    const float* nb2  = (const float*)d_in[31];
    const float* ng2  = (const float*)d_in[32];
    const float* nbt2 = (const float*)d_in[33];
    const float* cW1  = (const float*)d_in[34];
    const float* cb1  = (const float*)d_in[35];
    const float* cg1  = (const float*)d_in[36];
    const float* cbt1 = (const float*)d_in[37];
    const float* cW2  = (const float*)d_in[38];
    const float* cb2  = (const float*)d_in[39];

    // workspace layout (reused across branches)
    char* w = (char*)d_ws;
    size_t off = 0;
    auto alloc = [&](size_t bytes) -> void* {
        void* p = w + off;
        off = (off + bytes + 255) & ~(size_t)255;
        return p;
    };
    bf16_t* wt_e1   = (bf16_t*)alloc((size_t)128 * 416 * 2);
    bf16_t* wt_e2   = (bf16_t*)alloc((size_t)128 * 128 * 2);
    bf16_t* wt_n1   = (bf16_t*)alloc((size_t)128 * 256 * 2);
    bf16_t* wt_n2   = (bf16_t*)alloc((size_t)128 * 128 * 2);
    bf16_t* wt_c1   = (bf16_t*)alloc((size_t)128 * 128 * 2);
    bf16_t* edge_in = (bf16_t*)alloc((size_t)NE * 416 * 2);  // reused as node_in
    bf16_t* h1      = (bf16_t*)alloc((size_t)NE * 128 * 2);  // reused as ch1
    bf16_t* msg     = (bf16_t*)alloc((size_t)NE * 128 * 2);
    bf16_t* nh1     = (bf16_t*)alloc((size_t)NM * 128 * 2);
    float*  x_rel   = (float*)alloc((size_t)NE * 3 * 4);
    float*  msum    = (float*)alloc((size_t)NM * 128 * 4);
    float*  xsum    = (float*)alloc((size_t)NM * 3 * 4);
    float*  cnt     = (float*)alloc((size_t)NM * 4);

    // weights -> bf16 transposed, once (shared between branches)
    convert_wt_kernel<<<(128 * 416 + 255) / 256, 256, 0, stream>>>(eW1, wt_e1, 403, 416);
    convert_wt_kernel<<<(128 * 128 + 255) / 256, 256, 0, stream>>>(eW2, wt_e2, 128, 128);
    convert_wt_kernel<<<(128 * 256 + 255) / 256, 256, 0, stream>>>(nW1, wt_n1, 256, 256);
    convert_wt_kernel<<<(128 * 128 + 255) / 256, 256, 0, stream>>>(nW2, wt_n2, 128, 128);
    convert_wt_kernel<<<(128 * 128 + 255) / 256, 256, 0, stream>>>(cW1, wt_c1, 128, 128);

    for (int b = 0; b < 2; ++b) {
        float* out_h = (float*)d_out + (size_t)b * ((size_t)NM * 131);
        float* out_x = out_h + (size_t)NM * 128;

        hipMemsetAsync(msum, 0, (size_t)NM * 128 * 4, stream);
        hipMemsetAsync(xsum, 0, (size_t)NM * 3 * 4, stream);
        hipMemsetAsync(cnt,  0, (size_t)NM * 4, stream);

        edge_gather_kernel<<<(unsigned)(((size_t)NE * 416 + 255) / 256), 256, 0, stream>>>(
            fine_h[b], coarse_h[b], fine_x[b], pool_x[b], efeat[b],
            esrc[b], edst[b], edge_in, NE);
        xrel_kernel<<<(NE * 3 + 255) / 256, 256, 0, stream>>>(
            pool_x[b], esrc[b], edst[b], x_rel, NE);

        dim3 gE((NE + 127) / 128);
        // edge MLP L1: [E,416]x[416,128], LN+LReLU -> h1
        gemm_ln_kernel<416, false><<<gE, 256, 0, stream>>>(edge_in, NE, wt_e1,
            eb1, eg1, ebt1, 1, h1, nullptr, nullptr, 0.0f);
        // edge MLP L2: LN -> msg
        gemm_ln_kernel<128, false><<<gE, 256, 0, stream>>>(h1, NE, wt_e2,
            eb2, eg2, ebt2, 0, msg, nullptr, nullptr, 0.0f);
        // coords L1: LN+LReLU -> ch1 (reuse h1)
        gemm_ln_kernel<128, false><<<gE, 256, 0, stream>>>(msg, NE, wt_c1,
            cb1, cg1, cbt1, 1, h1, nullptr, nullptr, 0.0f);

        coef_xaggr_kernel<<<(NE + 255) / 256, 256, 0, stream>>>(
            h1, cW2, cb2, x_rel, edst[b], xsum, cnt, NE);
        msg_aggr_kernel<<<(unsigned)(((size_t)NE * 128 + 255) / 256), 256, 0, stream>>>(
            msg, edst[b], msum, NE);

        node_prep_kernel<<<(unsigned)(((size_t)NM * 256 + 255) / 256), 256, 0, stream>>>(
            pool_h[b], msum, cnt, edge_in /*node_in*/, NM);
        xout_kernel<<<(NM * 3 + 255) / 256, 256, 0, stream>>>(
            og_x[b], pool_x[b], xsum, cnt, out_x, NM);

        dim3 gN((NM + 127) / 128);
        // node MLP L1: [M,256]x[256,128], LN+LReLU -> nh1
        gemm_ln_kernel<256, false><<<gN, 256, 0, stream>>>(edge_in, NM, wt_n1,
            nb1, ng1, nbt1, 1, nh1, nullptr, nullptr, 0.0f);
        // node MLP L2: LN + skip (0.5*y + 0.5*pool_h) -> d_out h region (f32)
        gemm_ln_kernel<128, true><<<gN, 256, 0, stream>>>(nh1, NM, wt_n2,
            nb2, ng2, nbt2, 0, nullptr, out_h, pool_h[b], 0.5f);
    }
}
